// TemporalGCNBlock_71382356459943
// MI455X (gfx1250) — matrix operand order repaired and checked by reading, not compiled, not run on hardware
//
#include <hip/hip_runtime.h>
#include <stdint.h>

typedef _Float16 v16h __attribute__((ext_vector_type(16)));
typedef float    v8f  __attribute__((ext_vector_type(8)));

#define HDIM  128         // hidden size
#define PITCH 129         // padded LDS pitch in dwords (bank-conflict-free)
#define MPB   64          // rows (edges/nodes) per block = 4 M-tiles
#define MT    4           // M-tiles per block

// ---------------- setup: f32 -> packed f16 pairs ----------------
__global__ void k_cvt_x(const float* __restrict__ x, unsigned* __restrict__ x16u, int n) {
    int i = blockIdx.x * blockDim.x + threadIdx.x;
    if (i < n) {
        float2 v = ((const float2*)x)[i];
        union { _Float16 h[2]; unsigned u; } p;
        p.h[0] = (_Float16)v.x; p.h[1] = (_Float16)v.y;
        x16u[i] = p.u;
    }
}

// ---------------- setup: weight -> WMMA-B fragment order ----------------
// wf[((nt*8 + kk)*32 + lane)*16 + e] = W[32*kk + (lane>=16?8:0) + (e&7) + (e>=8?16:0)][nt*16 + lane%16]
__global__ void k_pack_w(const float* __restrict__ w, _Float16* __restrict__ wf) {
    int t = blockIdx.x * blockDim.x + threadIdx.x;   // 8*8*32*16 = 32768 total
    if (t >= 8 * 8 * 32 * 16) return;
    int e    =  t        & 15;
    int lane = (t >> 4)  & 31;
    int kk   = (t >> 9)  & 7;
    int nt   = (t >> 12) & 7;
    int K    = 32 * kk + ((lane & 16) ? 8 : 0) + (e & 7) + ((e & 8) ? 16 : 0);
    int col  = nt * 16 + (lane & 15);
    wf[t] = (_Float16)w[K * HDIM + col];
}

// A-fragment load from LDS for one 16-row M-tile, K-step kk
__device__ inline v16h load_A(const unsigned* lds, int mtrow, int kk, int khi2) {
    union { v16h h; unsigned u[8]; } A;
    int base = mtrow * PITCH + 16 * kk + khi2;
    #pragma unroll
    for (int d = 0; d < 4; ++d) {
        A.u[d]     = lds[base + d];      // K = 32kk + khi + {0..7}
        A.u[4 + d] = lds[base + 8 + d];  // K = 32kk + 16 + khi + {0..7}
    }
    return A.h;
}

// ---------------- edge message kernel: 64 edges / block ----------------
__launch_bounds__(256)
__global__ void k_edge(const unsigned* __restrict__ x16u,
                       const long long* __restrict__ ei,
                       const float* __restrict__ ew,
                       const v16h* __restrict__ wf,
                       const float* __restrict__ bb, const float* __restrict__ gg,
                       const float* __restrict__ bt,
                       float* __restrict__ msgs, float* __restrict__ counts,
                       int Ne) {
    __shared__ unsigned lds[MPB * PITCH];
    int t = threadIdx.x;
    int wave = t >> 5, lane = t & 31;
    int r = t >> 4, c = t & 15;
    long long ebase = (long long)blockIdx.x * MPB;

    // stage [x_src || x_dst] rows as f16 pairs
    #pragma unroll
    for (int mt = 0; mt < MT; ++mt) {
        long long eid = ebase + mt * 16 + r;
        long long ecl = (eid < Ne) ? eid : 0;
        int srcn = (int)ei[ecl];
        int dstn = (int)ei[(long long)Ne + ecl];
        #pragma unroll
        for (int i = 0; i < 8; ++i) {
            int col = i * 16 + c;
            unsigned v = (col < 64) ? x16u[(size_t)srcn * 64 + col]
                                    : x16u[(size_t)dstn * 64 + (col - 64)];
            lds[(mt * 16 + r) * PITCH + col] = v;
        }
    }
    __syncthreads();

    // B fragments resident in registers, reused across 4 M-tiles
    v16h Bf[8];
    #pragma unroll
    for (int kk = 0; kk < 8; ++kk) Bf[kk] = wf[(wave * 8 + kk) * 32 + lane];

    int arow = lane & 15;
    int khi2 = (lane & 16) ? 4 : 0;
    v8f acc[MT];
    #pragma unroll
    for (int mt = 0; mt < MT; ++mt) {
        acc[mt] = {};
        #pragma unroll
        for (int kk = 0; kk < 8; ++kk) {
            v16h A = load_A(lds, mt * 16 + arow, kk, khi2);
            acc[mt] = __builtin_amdgcn_wmma_f32_16x16x32_f16(false, A, false, Bf[kk],
                                                             (short)0, acc[mt], false, false);
        }
    }

    // per M-tile: spill C, LayerNorm+ReLU+weight, scatter-add
    float* outf = (float*)lds;
    int mlo  = (lane & 16) ? 8 : 0;
    int ccol = wave * 16 + (lane & 15);
    #pragma unroll
    for (int mt = 0; mt < MT; ++mt) {
        __syncthreads();
        #pragma unroll
        for (int v = 0; v < 8; ++v) outf[(v + mlo) * PITCH + ccol] = acc[mt][v];
        __syncthreads();

        long long eid = ebase + mt * 16 + r;
        int okr = eid < Ne;
        long long ecl = okr ? eid : 0;
        int dstn  = (int)ei[(long long)Ne + ecl];
        float w_e = okr ? ew[ecl] : 0.0f;

        float vv[8], s = 0.f, s2 = 0.f;
        #pragma unroll
        for (int i = 0; i < 8; ++i) {
            int col = i * 16 + c;
            float v = outf[r * PITCH + col] + bb[col];
            vv[i] = v; s += v; s2 += v * v;
        }
        #pragma unroll
        for (int m = 8; m >= 1; m >>= 1) {
            s  += __shfl_xor(s,  m, 32);
            s2 += __shfl_xor(s2, m, 32);
        }
        float mu  = s * (1.f / HDIM);
        float var = s2 * (1.f / HDIM) - mu * mu;
        float rs  = rsqrtf(var + 1e-5f);
        if (okr) {
            #pragma unroll
            for (int i = 0; i < 8; ++i) {
                int col = i * 16 + c;
                float v = (vv[i] - mu) * rs * gg[col] + bt[col];
                v = v > 0.f ? v : 0.f;
                atomicAdd(&msgs[(size_t)dstn * HDIM + col], v * w_e);
            }
            if (c == 0) atomicAdd(&counts[dstn], 1.0f);
        }
    }
}

// ---------------- node update kernel: 64 nodes / block ----------------
__launch_bounds__(256)
__global__ void k_update(const float* __restrict__ x,
                         const unsigned* __restrict__ x16u,
                         const float* __restrict__ msgs, const float* __restrict__ counts,
                         const v16h* __restrict__ wf,
                         const float* __restrict__ bb, const float* __restrict__ gg,
                         const float* __restrict__ bt,
                         const float* __restrict__ gw, const float* __restrict__ gb,
                         float* __restrict__ out, int Nn) {
    __shared__ unsigned lds[MPB * PITCH];
    int t = threadIdx.x;
    int wave = t >> 5, lane = t & 31;
    int r = t >> 4, c = t & 15;
    int nbase = blockIdx.x * MPB;

    // stage [x || normalized messages] as f16 pairs
    #pragma unroll
    for (int mt = 0; mt < MT; ++mt) {
        int n  = nbase + mt * 16 + r;
        int nn = (n < Nn) ? n : 0;
        float cnt  = counts[nn];
        float minv = (cnt > 0.f) ? 1.f / (cnt + 1e-8f) : 0.f;
        #pragma unroll
        for (int i = 0; i < 8; ++i) {
            int col = i * 16 + c;
            unsigned v;
            if (col < 64) {
                v = x16u[(size_t)nn * 64 + col];
            } else {
                int j2 = (col - 64) * 2;
                float m0 = msgs[(size_t)nn * HDIM + j2]     * minv;
                float m1 = msgs[(size_t)nn * HDIM + j2 + 1] * minv;
                union { _Float16 h[2]; unsigned u; } p;
                p.h[0] = (_Float16)m0; p.h[1] = (_Float16)m1;
                v = p.u;
            }
            lds[(mt * 16 + r) * PITCH + col] = v;
        }
    }
    __syncthreads();

    v16h Bf[8];
    #pragma unroll
    for (int kk = 0; kk < 8; ++kk) Bf[kk] = wf[(wave * 8 + kk) * 32 + lane];

    int arow = lane & 15;
    int khi2 = (lane & 16) ? 4 : 0;
    v8f acc[MT];
    #pragma unroll
    for (int mt = 0; mt < MT; ++mt) {
        acc[mt] = {};
        #pragma unroll
        for (int kk = 0; kk < 8; ++kk) {
            v16h A = load_A(lds, mt * 16 + arow, kk, khi2);
            acc[mt] = __builtin_amdgcn_wmma_f32_16x16x32_f16(false, A, false, Bf[kk],
                                                             (short)0, acc[mt], false, false);
        }
    }

    float* outf = (float*)lds;
    int mlo  = (lane & 16) ? 8 : 0;
    int ccol = wave * 16 + (lane & 15);
    #pragma unroll
    for (int mt = 0; mt < MT; ++mt) {
        __syncthreads();
        #pragma unroll
        for (int v = 0; v < 8; ++v) outf[(v + mlo) * PITCH + ccol] = acc[mt][v];
        __syncthreads();

        int n  = nbase + mt * 16 + r;
        int ok = n < Nn;
        int nn = ok ? n : 0;

        float vv[8], xv[8], s = 0.f, s2 = 0.f, gp = 0.f;
        #pragma unroll
        for (int i = 0; i < 8; ++i) {
            int col = i * 16 + c;
            float xval = x[(size_t)nn * HDIM + col];
            xv[i] = xval;
            gp += xval * gw[col];
            float v = outf[r * PITCH + col] + bb[col];
            vv[i] = v; s += v; s2 += v * v;
        }
        #pragma unroll
        for (int m = 8; m >= 1; m >>= 1) {
            s  += __shfl_xor(s,  m, 32);
            s2 += __shfl_xor(s2, m, 32);
            gp += __shfl_xor(gp, m, 32);
        }
        float mu   = s * (1.f / HDIM);
        float var  = s2 * (1.f / HDIM) - mu * mu;
        float rs   = rsqrtf(var + 1e-5f);
        float gate = 1.f / (1.f + __expf(-(gp + gb[0])));
        if (ok) {
            #pragma unroll
            for (int i = 0; i < 8; ++i) {
                int col = i * 16 + c;
                float h = (vv[i] - mu) * rs * gg[col] + bt[col];
                h = h > 0.f ? h : 0.f;
                out[(size_t)n * HDIM + col] = gate * h + (1.f - gate) * xv[i];
            }
        }
    }
}

extern "C" void kernel_launch(void* const* d_in, const int* in_sizes, int n_in,
                              void* d_out, int out_size, void* d_ws, size_t ws_size,
                              hipStream_t stream) {
    (void)n_in; (void)out_size; (void)ws_size;
    const float*     x        = (const float*)d_in[0];
    const long long* ei       = (const long long*)d_in[1];
    const float*     ew       = (const float*)d_in[2];
    const float*     msg_w    = (const float*)d_in[3];
    const float*     msg_b    = (const float*)d_in[4];
    const float*     msg_g    = (const float*)d_in[5];
    const float*     msg_beta = (const float*)d_in[6];
    const float*     upd_w    = (const float*)d_in[7];
    const float*     upd_b    = (const float*)d_in[8];
    const float*     upd_g    = (const float*)d_in[9];
    const float*     upd_beta = (const float*)d_in[10];
    const float*     gate_w   = (const float*)d_in[11];
    const float*     gate_b   = (const float*)d_in[12];

    int Nn = in_sizes[0] / HDIM;   // nodes
    int Ne = in_sizes[2];          // edges

    char* ws = (char*)d_ws;
    size_t off = 0;
    auto take = [&](size_t bytes) -> char* {
        char* p = ws + off;
        off += (bytes + 255) & ~(size_t)255;
        return p;
    };
    float*    msgs   = (float*)take((size_t)Nn * HDIM * sizeof(float));
    float*    counts = (float*)take((size_t)Nn * sizeof(float));
    unsigned* x16u   = (unsigned*)take((size_t)Nn * 64 * sizeof(unsigned));
    _Float16* msg_wf = (_Float16*)take((size_t)256 * 128 * sizeof(_Float16));
    _Float16* upd_wf = (_Float16*)take((size_t)256 * 128 * sizeof(_Float16));

    hipMemsetAsync(msgs,   0, (size_t)Nn * HDIM * sizeof(float), stream);
    hipMemsetAsync(counts, 0, (size_t)Nn * sizeof(float), stream);

    int ncvt = Nn * 64;
    k_cvt_x<<<(ncvt + 255) / 256, 256, 0, stream>>>(x, x16u, ncvt);
    k_pack_w<<<128, 256, 0, stream>>>(msg_w, msg_wf);
    k_pack_w<<<128, 256, 0, stream>>>(upd_w, upd_wf);

    k_edge<<<(Ne + MPB - 1) / MPB, 256, 0, stream>>>(
        x16u, ei, ew, (const v16h*)msg_wf, msg_b, msg_g, msg_beta,
        msgs, counts, Ne);

    k_update<<<(Nn + MPB - 1) / MPB, 256, 0, stream>>>(
        x, x16u, msgs, counts, (const v16h*)upd_wf, upd_b, upd_g, upd_beta,
        gate_w, gate_b, (float*)d_out, Nn);
}
